// SparseAutoencoder_33028298506893
// MI455X (gfx1250) — matrix operand order repaired and checked by reading, not compiled, not run on hardware
//
#include <hip/hip_runtime.h>

typedef __attribute__((ext_vector_type(16))) __bf16 v16bf;
typedef __attribute__((ext_vector_type(8)))  __bf16 v8bf;
typedef __attribute__((ext_vector_type(4)))  __bf16 v4bf;
typedef __attribute__((ext_vector_type(8)))  float  v8f;

static constexpr int BATCH = 4096;
static constexpr int DIM   = 768;     // K of encoder GEMM
static constexpr int LAT   = 16384;   // N of encoder GEMM
static constexpr int TOPK  = 32;

// ---------------------------------------------------------------------------
// Kernel 1: split fp32 -> bf16 hi + bf16 lo residual  (3-term split GEMM prep)
// ---------------------------------------------------------------------------
__global__ __launch_bounds__(256)
void pack_hi_lo(const float* __restrict__ in, __bf16* __restrict__ hi,
                __bf16* __restrict__ lo, int n) {
  int i = (blockIdx.x * 256 + threadIdx.x) * 4;
  if (i >= n) return;
  float4 v = *reinterpret_cast<const float4*>(in + i);
  float f[4] = {v.x, v.y, v.z, v.w};
  v4bf h, l;
#pragma unroll
  for (int c = 0; c < 4; ++c) {
    __bf16 hb = (__bf16)f[c];
    h[c] = hb;
    l[c] = (__bf16)(f[c] - (float)hb);
  }
  *reinterpret_cast<v4bf*>(hi + i) = h;
  *reinterpret_cast<v4bf*>(lo + i) = l;
}

// ---------------------------------------------------------------------------
// Kernel 2: transpose dec_w [768, 16384] -> dec_wT [16384, 768]
// ---------------------------------------------------------------------------
__global__ __launch_bounds__(256)
void transpose_dw(const float* __restrict__ in, float* __restrict__ outT) {
  __shared__ float tile[32][33];
  int xl = blockIdx.x * 32 + threadIdx.x;   // LAT index
  int yd = blockIdx.y * 32 + threadIdx.y;   // DIM index
#pragma unroll
  for (int j = 0; j < 4; ++j)
    tile[threadIdx.y + j * 8][threadIdx.x] = in[(size_t)(yd + j * 8) * LAT + xl];
  __syncthreads();
  int xd = blockIdx.y * 32 + threadIdx.x;   // DIM
  int yl = blockIdx.x * 32 + threadIdx.y;   // LAT
#pragma unroll
  for (int j = 0; j < 4; ++j)
    outT[(size_t)(yl + j * 8) * DIM + xd] = tile[threadIdx.x][threadIdx.y + j * 8];
}

// ---------------------------------------------------------------------------
// Kernel 3: encoder GEMM   pre[B,L] = X[B,K] * W[L,K]^T + bias
// bf16 3-term split, V_WMMA_F32_16X16X32_BF16, fp32 accumulate.
// Double-buffered LDS; A tiles staged with GLOBAL_LOAD_ASYNC_TO_LDS_B128
// (ASYNCcnt), B tiles transposed-on-store; one barrier per K-step.
// ---------------------------------------------------------------------------
__global__ __launch_bounds__(256)
void enc_gemm(const __bf16* __restrict__ Xhi, const __bf16* __restrict__ Xlo,
              const __bf16* __restrict__ Whi, const __bf16* __restrict__ Wlo,
              const float* __restrict__ bias, float* __restrict__ out) {
  constexpr int K   = DIM;
  constexpr int NT  = K / 32;    // 24 K-steps
  constexpr int BNP = 136;       // padded N-stride of transposed B tiles
  __shared__ __align__(16) __bf16 sAhi[2][128 * 32];
  __shared__ __align__(16) __bf16 sAlo[2][128 * 32];
  __shared__ __align__(16) __bf16 sBhi[2][32 * BNP];
  __shared__ __align__(16) __bf16 sBlo[2][32 * BNP];

  const int tid  = threadIdx.x;
  const int lane = tid & 31;
  const int wv   = tid >> 5;     // wave 0..7
  const int wm   = wv >> 2;      // 0..1  -> M offset wm*64
  const int wn   = wv & 3;       // 0..3  -> N offset wn*32
  const int m0   = blockIdx.y * 128;
  const int n0   = blockIdx.x * 128;

  // A staging geometry: 512 chunks of 8 bf16 (16B), 2 per thread
  const int rowA0 = (tid * 2) >> 2, kcA0 = ((tid * 2) & 3) * 8;
  const int rowA1 = (tid * 2 + 1) >> 2, kcA1 = ((tid * 2 + 1) & 3) * 8;
  // B staging geometry
  const int bn  = tid & 127;     // W row within N tile
  const int bkh = tid >> 7;      // K half (0 -> K 0..15, 1 -> K 16..31)

  // Issue async memory->LDS copies of one A tile (hi+lo), 16B per chunk.
  auto stageA_async = [&](int buf, int k0) {
    unsigned lh0 = (unsigned)(uintptr_t)(&sAhi[buf][rowA0 * 32 + kcA0]);
    unsigned lh1 = (unsigned)(uintptr_t)(&sAhi[buf][rowA1 * 32 + kcA1]);
    unsigned ll0 = (unsigned)(uintptr_t)(&sAlo[buf][rowA0 * 32 + kcA0]);
    unsigned ll1 = (unsigned)(uintptr_t)(&sAlo[buf][rowA1 * 32 + kcA1]);
    const __bf16* gh0 = Xhi + (size_t)(m0 + rowA0) * K + k0 + kcA0;
    const __bf16* gh1 = Xhi + (size_t)(m0 + rowA1) * K + k0 + kcA1;
    const __bf16* gl0 = Xlo + (size_t)(m0 + rowA0) * K + k0 + kcA0;
    const __bf16* gl1 = Xlo + (size_t)(m0 + rowA1) * K + k0 + kcA1;
    asm volatile("global_load_async_to_lds_b128 %0, %1, off"
                 :: "v"(lh0), "v"(gh0) : "memory");
    asm volatile("global_load_async_to_lds_b128 %0, %1, off"
                 :: "v"(lh1), "v"(gh1) : "memory");
    asm volatile("global_load_async_to_lds_b128 %0, %1, off"
                 :: "v"(ll0), "v"(gl0) : "memory");
    asm volatile("global_load_async_to_lds_b128 %0, %1, off"
                 :: "v"(ll1), "v"(gl1) : "memory");
  };
  // Load one B tile row-chunk into registers (coalesced 32B per array).
  auto stageB_load = [&](int k0, __bf16 th[16], __bf16 tl[16]) {
    const size_t g = (size_t)(n0 + bn) * K + k0 + bkh * 16;
    *reinterpret_cast<uint4*>(th)     = *reinterpret_cast<const uint4*>(Whi + g);
    *reinterpret_cast<uint4*>(th + 8) = *reinterpret_cast<const uint4*>(Whi + g + 8);
    *reinterpret_cast<uint4*>(tl)     = *reinterpret_cast<const uint4*>(Wlo + g);
    *reinterpret_cast<uint4*>(tl + 8) = *reinterpret_cast<const uint4*>(Wlo + g + 8);
  };
  // Scatter B registers into LDS transposed [k][n] (lane==N fragment layout).
  auto stageB_store = [&](int buf, const __bf16 th[16], const __bf16 tl[16]) {
#pragma unroll
    for (int i = 0; i < 16; ++i) {
      sBhi[buf][(bkh * 16 + i) * BNP + bn] = th[i];
      sBlo[buf][(bkh * 16 + i) * BNP + bn] = tl[i];
    }
  };

  v8f acc[4][2];
#pragma unroll
  for (int i = 0; i < 4; ++i)
#pragma unroll
    for (int j = 0; j < 2; ++j)
#pragma unroll
      for (int e = 0; e < 8; ++e) acc[i][j][e] = 0.0f;

  // ---- prologue: fill buffer 0 ----
  {
    stageA_async(0, 0);
    __bf16 th[16], tl[16];
    stageB_load(0, th, tl);
    stageB_store(0, th, tl);
    asm volatile("s_wait_asynccnt 0x0" ::: "memory");
    __syncthreads();
  }

  for (int kt = 0; kt < NT; ++kt) {
    const int cur = kt & 1, nxt = cur ^ 1;
    const bool hasNext = (kt + 1) < NT;

    __bf16 th[16], tl[16];
    if (hasNext) {
      stageB_load((kt + 1) * 32, th, tl);   // global loads in flight
      stageA_async(nxt, (kt + 1) * 32);     // async fills overlap compute
    }

    // ---- load fragments from current buffer ----
    // A 16-bit layout: lane<16 -> row M=lane, K {kb..kb+7, kb+16..kb+23}, kb=0
    //                  lane>=16 -> same row,  kb=8
    v16bf ah[4], al[4];
#pragma unroll
    for (int i = 0; i < 4; ++i) {
      int mrow = wm * 64 + i * 16 + (lane & 15);
      int kb   = (lane >> 4) * 8;
      const v8bf c0h = *reinterpret_cast<const v8bf*>(&sAhi[cur][mrow * 32 + kb]);
      const v8bf c1h = *reinterpret_cast<const v8bf*>(&sAhi[cur][mrow * 32 + kb + 16]);
      const v8bf c0l = *reinterpret_cast<const v8bf*>(&sAlo[cur][mrow * 32 + kb]);
      const v8bf c1l = *reinterpret_cast<const v8bf*>(&sAlo[cur][mrow * 32 + kb + 16]);
#pragma unroll
      for (int e = 0; e < 8; ++e) {
        ah[i][e] = c0h[e]; ah[i][8 + e] = c1h[e];
        al[i][e] = c0l[e]; al[i][8 + e] = c1l[e];
      }
    }
    // B layout: lane -> N = lane%16; element s -> K = 8*(lane/16) + s%8 + 16*(s/8)
    v16bf bh[2], bl[2];
#pragma unroll
    for (int j = 0; j < 2; ++j) {
      int nb = wn * 32 + j * 16;
#pragma unroll
      for (int s = 0; s < 16; ++s) {
        int kk = (lane >> 4) * 8 + (s & 7) + (s >> 3) * 16;
        bh[j][s] = sBhi[cur][kk * BNP + nb + (lane & 15)];
        bl[j][s] = sBlo[cur][kk * BNP + nb + (lane & 15)];
      }
    }

    // ---- 3-term split MMA: hi*hi + hi*lo + lo*hi ----
#pragma unroll
    for (int i = 0; i < 4; ++i)
#pragma unroll
      for (int j = 0; j < 2; ++j) {
        acc[i][j] = __builtin_amdgcn_wmma_f32_16x16x32_bf16(
            false, ah[i], false, bh[j], (short)0, acc[i][j], false, false);
        acc[i][j] = __builtin_amdgcn_wmma_f32_16x16x32_bf16(
            false, ah[i], false, bl[j], (short)0, acc[i][j], false, false);
        acc[i][j] = __builtin_amdgcn_wmma_f32_16x16x32_bf16(
            false, al[i], false, bh[j], (short)0, acc[i][j], false, false);
      }

    if (hasNext) stageB_store(nxt, th, tl);
    asm volatile("s_wait_asynccnt 0x0" ::: "memory");
    __syncthreads();
  }

  // ---- epilogue: C layout lane<16 -> N=lane, M=r ; lane>=16 -> N=lane-16, M=r+8
#pragma unroll
  for (int j = 0; j < 2; ++j) {
    int col = n0 + wn * 32 + j * 16 + (lane & 15);
    float bv = bias[col];
#pragma unroll
    for (int i = 0; i < 4; ++i) {
      int mbase = m0 + wm * 64 + i * 16 + ((lane >> 4) ? 8 : 0);
#pragma unroll
      for (int r = 0; r < 8; ++r)
        out[(size_t)(mbase + r) * LAT + col] = acc[i][j][r] + bv;
    }
  }
}

// ---------------------------------------------------------------------------
// Kernel 4: exact top-32 by |pre| per row (1 block/row, 256 threads).
// Register-resident 64 values/thread; 32 strict-decreasing extraction passes
// with (value, index) total order => exactly-k, index tie-break like lax.top_k.
// ---------------------------------------------------------------------------
__global__ __launch_bounds__(256)
void topk_kernel(const float* __restrict__ pre, float* __restrict__ latents,
                 float* __restrict__ topv, int* __restrict__ topi) {
  const int row = blockIdx.x;
  const int tid = threadIdx.x;
  const int lane = tid & 31;
  const int wv = tid >> 5;
  const float* prow = pre + (size_t)row * LAT;

  __shared__ int   flagw[LAT / 4];
  __shared__ float swv[8];
  __shared__ int   swi[8];
  __shared__ int   li[TOPK];
  __shared__ float s_selv;
  __shared__ int   s_seli;
  unsigned char* flag = reinterpret_cast<unsigned char*>(flagw);

#pragma unroll
  for (int s = 0; s < 16; ++s) flagw[tid + (s << 8)] = 0;

  float av[64];
#pragma unroll
  for (int s = 0; s < 64; ++s) av[s] = fabsf(prow[tid + (s << 8)]);

  float pv = __builtin_inff();
  int   pi = -1;

  for (int p = 0; p < TOPK; ++p) {
    float bm = -1.0f;
    int   bi = 0x7fffffff;
#pragma unroll
    for (int s = 0; s < 64; ++s) {
      int idx = tid + (s << 8);
      float v = av[s];
      bool less_prev = (v < pv) || (v == pv && idx > pi);
      bool better    = (v > bm) || (v == bm && idx < bi);
      if (less_prev && better) { bm = v; bi = idx; }
    }
#pragma unroll
    for (int off = 16; off > 0; off >>= 1) {
      float ov = __shfl_xor(bm, off, 32);
      int   oi = __shfl_xor(bi, off, 32);
      if (ov > bm || (ov == bm && oi < bi)) { bm = ov; bi = oi; }
    }
    if (lane == 0) { swv[wv] = bm; swi[wv] = bi; }
    __syncthreads();
    if (tid == 0) {
      float gv = swv[0]; int gi = swi[0];
#pragma unroll
      for (int w = 1; w < 8; ++w)
        if (swv[w] > gv || (swv[w] == gv && swi[w] < gi)) { gv = swv[w]; gi = swi[w]; }
      s_selv = gv; s_seli = gi;
      li[p] = gi;
      topv[row * TOPK + p] = prow[gi];
      topi[row * TOPK + p] = gi;
    }
    __syncthreads();
    pv = s_selv; pi = s_seli;
  }

  if (tid < TOPK) flag[li[tid]] = 1;
  __syncthreads();

  float* lrow = latents + (size_t)row * LAT;
#pragma unroll
  for (int s = 0; s < 64; ++s) {
    int i = tid + (s << 8);
    lrow[i] = flag[i] ? prow[i] : 0.0f;
  }
}

// ---------------------------------------------------------------------------
// Kernel 5: sparse decoder  recon[b,:] = sum_j v_j * dec_wT[idx_j,:] + dec_b
// ---------------------------------------------------------------------------
__global__ __launch_bounds__(256)
void dec_sparse(const float* __restrict__ topv, const int* __restrict__ topi,
                const float* __restrict__ dwT, const float* __restrict__ bias,
                float* __restrict__ recon) {
  const int row = blockIdx.x;
  const int tid = threadIdx.x;
  __shared__ float sv[TOPK];
  __shared__ int   si[TOPK];
  if (tid < TOPK) { sv[tid] = topv[row * TOPK + tid]; si[tid] = topi[row * TOPK + tid]; }
  __syncthreads();
#pragma unroll
  for (int c = 0; c < 3; ++c) {
    int d = tid + (c << 8);
    float acc = bias[d];
#pragma unroll 8
    for (int j = 0; j < TOPK; ++j) acc += sv[j] * dwT[(size_t)si[j] * DIM + d];
    recon[(size_t)row * DIM + d] = acc;
  }
}

// ---------------------------------------------------------------------------
extern "C" void kernel_launch(void* const* d_in, const int* in_sizes, int n_in,
                              void* d_out, int out_size, void* d_ws, size_t ws_size,
                              hipStream_t stream) {
  const float* x     = (const float*)d_in[0];
  const float* enc_w = (const float*)d_in[1];
  const float* enc_b = (const float*)d_in[2];
  const float* dec_w = (const float*)d_in[3];
  const float* dec_b = (const float*)d_in[4];

  float* out     = (float*)d_out;
  float* recon   = out;                                   // [4096, 768]
  float* latents = out + (size_t)BATCH * DIM;             // [4096, 16384]
  float* pre     = latents + (size_t)BATCH * LAT;         // [4096, 16384]

  char* ws = (char*)d_ws;
  size_t o = 0;
  __bf16* Xhi = (__bf16*)(ws + o); o += (size_t)BATCH * DIM * 2;
  __bf16* Xlo = (__bf16*)(ws + o); o += (size_t)BATCH * DIM * 2;
  __bf16* Whi = (__bf16*)(ws + o); o += (size_t)LAT * DIM * 2;
  __bf16* Wlo = (__bf16*)(ws + o); o += (size_t)LAT * DIM * 2;
  float*  dwT = (float*)(ws + o);  o += (size_t)LAT * DIM * 4;
  float*  tv  = (float*)(ws + o);  o += (size_t)BATCH * TOPK * 4;
  int*    ti  = (int*)(ws + o);    o += (size_t)BATCH * TOPK * 4;

  pack_hi_lo<<<(BATCH * DIM) / 1024, 256, 0, stream>>>(x, Xhi, Xlo, BATCH * DIM);
  pack_hi_lo<<<(LAT * DIM) / 1024, 256, 0, stream>>>(enc_w, Whi, Wlo, LAT * DIM);
  transpose_dw<<<dim3(LAT / 32, DIM / 32), dim3(32, 8), 0, stream>>>(dec_w, dwT);
  enc_gemm<<<dim3(LAT / 128, BATCH / 128), 256, 0, stream>>>(Xhi, Xlo, Whi, Wlo,
                                                             enc_b, pre);
  topk_kernel<<<BATCH, 256, 0, stream>>>(pre, latents, tv, ti);
  dec_sparse<<<BATCH, 256, 0, stream>>>(tv, ti, dwT, dec_b, recon);
}
